// RoIBBox_2345052143695
// MI455X (gfx1250) — compile-verified, matrix operand order checked by reading, not run on hardware
//
#include <hip/hip_runtime.h>
#include <hip/hip_bf16.h>

#define BATCH 32
#define ANUM  90000
#define NGT   20
#define TOPN  300
#define NPOS  64
#define NEG_INF_F (-1.0e9f)

typedef __attribute__((ext_vector_type(16))) _Float16 v16h;
typedef __attribute__((ext_vector_type(8)))  float    v8f;

// ---------------------------------------------------------------------------
// Kernel 1: decode anchors + deltas -> clipped boxes, copy scores (mutable)
// ---------------------------------------------------------------------------
__global__ void decode_kernel(const float* __restrict__ anchors,
                              const float* __restrict__ deltas,
                              const float* __restrict__ labels,
                              float* __restrict__ boxes,
                              float* __restrict__ scores) {
    int idx = blockIdx.x * blockDim.x + threadIdx.x;
    if (idx >= BATCH * ANUM) return;
    float4 a = ((const float4*)anchors)[idx];
    float4 d = ((const float4*)deltas)[idx];
    float ah = a.z - a.x;
    float aw = a.w - a.y;
    float acy = a.x + 0.5f * ah;
    float acx = a.y + 0.5f * aw;
    float h = __expf(d.z) * ah;
    float w = __expf(d.w) * aw;
    float cy = d.x * ah + acy;
    float cx = d.y * aw + acx;
    float4 b;
    b.x = fminf(fmaxf(cy - 0.5f * h, 0.f), 1.f);
    b.y = fminf(fmaxf(cx - 0.5f * w, 0.f), 1.f);
    b.z = fminf(fmaxf(cy + 0.5f * h, 0.f), 1.f);
    b.w = fminf(fmaxf(cx + 0.5f * w, 0.f), 1.f);
    ((float4*)boxes)[idx] = b;
    scores[idx] = labels[idx];
}

// ---------------------------------------------------------------------------
// Kernel 2: greedy NMS, one workgroup (32 wave32 waves) per batch.
// Cached per-thread local max: rescan strip only if own argmax got suppressed.
// ---------------------------------------------------------------------------
__global__ __launch_bounds__(1024) void nms_kernel(const float* __restrict__ boxes,
                                                   float* __restrict__ scores,
                                                   float* __restrict__ nms_out) {
    const int b = blockIdx.x;
    const float* bx = boxes + (size_t)b * ANUM * 4;
    float* sc = scores + (size_t)b * ANUM;
    float* out = nms_out + (size_t)b * TOPN * 4;
    const int tid  = threadIdx.x;
    const int lane = tid & 31;
    const int wave = tid >> 5;

    __shared__ float s_wval[32];
    __shared__ int   s_widx[32];
    __shared__ float s_box[4];

    float lbest = -__builtin_inff();
    int   lidx  = 0;
    for (int j = tid; j < ANUM; j += 1024) {
        float v = sc[j];
        if (v > lbest) { lbest = v; lidx = j; }
    }

    for (int it = 0; it < TOPN; ++it) {
        // wave32 butterfly argmax with jnp.argmax tie-break (first max)
        float v = lbest; int id = lidx;
        #pragma unroll
        for (int off = 16; off >= 1; off >>= 1) {
            float ov = __shfl_xor(v, off, 32);
            int   oi = __shfl_xor(id, off, 32);
            if (ov > v || (ov == v && oi < id)) { v = ov; id = oi; }
        }
        if (lane == 0) { s_wval[wave] = v; s_widx[wave] = id; }
        __syncthreads();
        if (wave == 0) {
            float v2 = s_wval[lane]; int i2 = s_widx[lane];
            #pragma unroll
            for (int off = 16; off >= 1; off >>= 1) {
                float ov = __shfl_xor(v2, off, 32);
                int   oi = __shfl_xor(i2, off, 32);
                if (ov > v2 || (ov == v2 && oi < i2)) { v2 = ov; i2 = oi; }
            }
            if (lane == 0) {
                float4 bb = ((const float4*)bx)[i2];
                s_box[0] = bb.x; s_box[1] = bb.y; s_box[2] = bb.z; s_box[3] = bb.w;
                bool valid = v2 > (NEG_INF_F * 0.5f);
                float4 ob = valid ? bb : float4{0.f, 0.f, 0.f, 0.f};
                ((float4*)out)[it] = ob;
            }
        }
        __syncthreads();
        const float by1 = s_box[0], bx1 = s_box[1], by2 = s_box[2], bx2 = s_box[3];
        const float barea = (by2 - by1) * (bx2 - bx1);
        bool lost = false;
        for (int j = tid; j < ANUM; j += 1024) {
            float4 c = ((const float4*)bx)[j];
            float yy1 = fmaxf(by1, c.x), xx1 = fmaxf(bx1, c.y);
            float yy2 = fminf(by2, c.z), xx2 = fminf(bx2, c.w);
            float inter = fmaxf(yy2 - yy1, 0.f) * fmaxf(xx2 - xx1, 0.f);
            float carea = (c.z - c.x) * (c.w - c.y);
            float uni   = barea + carea - inter;
            float iou   = (uni > 0.f) ? inter / uni : 0.f;
            if (iou > 0.5f) {
                sc[j] = NEG_INF_F;
                if (j == lidx) lost = true;
            }
        }
        if (lost) {  // scores only decrease -> rescan only when needed
            lbest = -__builtin_inff(); lidx = 0;
            for (int j = tid; j < ANUM; j += 1024) {
                float vv = sc[j];
                if (vv > lbest) { lbest = vv; lidx = j; }
            }
        }
        __syncthreads();
    }
}

// ---------------------------------------------------------------------------
// Kernel 3: IoU of 300 NMS boxes vs 20 GT boxes; max + first-argmax over GT
// ---------------------------------------------------------------------------
__global__ void iou_gt_kernel(const float* __restrict__ nmsb,
                              const float* __restrict__ gt,
                              float* __restrict__ merged,
                              int* __restrict__ maxgt) {
    const int b = blockIdx.x;
    const int i = threadIdx.x;
    __shared__ float4 sgt[NGT];
    if (i < NGT) sgt[i] = ((const float4*)gt)[b * NGT + i];
    __syncthreads();
    if (i >= TOPN) return;
    float4 nb = ((const float4*)nmsb)[b * TOPN + i];
    float a1 = (nb.z - nb.x) * (nb.w - nb.y);
    float bestv = -__builtin_inff();
    int   bestj = 0;
    for (int j = 0; j < NGT; ++j) {
        float4 g = sgt[j];
        float y1 = fmaxf(nb.x, g.x), x1 = fmaxf(nb.y, g.y);
        float y2 = fminf(nb.z, g.z), x2 = fminf(nb.w, g.w);
        float inter = fmaxf(y2 - y1, 0.f) * fmaxf(x2 - x1, 0.f);
        float a2 = (g.z - g.x) * (g.w - g.y);
        float uni = a1 + a2 - inter;
        float iou = (uni > 0.f) ? inter / uni : 0.f;
        if (iou > bestv) { bestv = iou; bestj = j; }   // first max
    }
    merged[b * TOPN + i] = bestv;
    maxgt[b * TOPN + i]  = bestj;
}

// ---------------------------------------------------------------------------
// Kernel 4: stable-descending-sort ranks via WMMA row-sum of the comparison
// matrix: rank(i) = sum_j [ m_j > m_i || (m_j == m_i && j < i) ].
// Branchless compare bits in f16 (exactly 0/1), B = ones, f32 accumulation
// (sums <= 300 -> exact). Pad sentinel -1 loses every compare (merged >= 0).
// grid (19 row-tiles, 32 batches), 1 wave per block (EXEC all-ones at WMMA).
// ---------------------------------------------------------------------------
__global__ __launch_bounds__(32) void rank_kernel(const float* __restrict__ merged,
                                                  int* __restrict__ rank) {
    const int tile = blockIdx.x;          // 0..18  (19*16 = 304 >= 300 rows)
    const int b    = blockIdx.y;          // 0..31
    const int lane = threadIdx.x;         // 0..31
    const float* m = merged + b * TOPN;
    __shared__ __align__(16) float sm[320];
    for (int j = lane; j < 320; j += 32) sm[j] = (j < TOPN) ? m[j] : -1.0f;
    __syncthreads();

    const int hi = lane >> 4;             // half-wave selects K-subblock
    const int i  = tile * 16 + (lane & 15);
    const float mi = sm[i];               // pad rows -> garbage, discarded

    v16h ones;
    #pragma unroll
    for (int e = 0; e < 16; ++e) ones[e] = (_Float16)1.0f;

    const float4* smv = (const float4*)sm;
    v8f acc = {};
    #pragma unroll
    for (int kc = 0; kc < 10; ++kc) {     // K = 10 * 32 = 320 columns
        const int fb = kc * 8 + 2 * hi;   // float4 index of sm[kc*32 + 8*hi]
        float4 q0 = smv[fb];
        float4 q1 = smv[fb + 1];
        float4 q2 = smv[fb + 4];          // sm[kc*32 + 8*hi + 16]
        float4 q3 = smv[fb + 5];
        float mj[16] = {q0.x, q0.y, q0.z, q0.w, q1.x, q1.y, q1.z, q1.w,
                        q2.x, q2.y, q2.z, q2.w, q3.x, q3.y, q3.z, q3.w};
        const int j0 = kc * 32 + 8 * hi;
        v16h a;
        #pragma unroll
        for (int e = 0; e < 16; ++e) {
            int j = j0 + ((e < 8) ? e : e + 8);
            float v = mj[e];
            // non-short-circuit -> pure v_cmp / v_cndmask, no exec branching
            int c = (int)(v > mi) | ((int)(v == mi) & (int)(j < i));
            a[e] = (_Float16)(float)c;
        }
        acc = __builtin_amdgcn_wmma_f32_16x16x32_f16(
            /*neg_a=*/false, a, /*neg_b=*/false, ones,
            /*c_mod=*/(short)0, acc, /*reuse_a=*/false, /*reuse_b=*/false);
    }
    // D layout: VGPR r holds row M = r + 8*hi, replicated across 16 columns.
    if ((lane & 15) == 0) {
        #pragma unroll
        for (int r = 0; r < 8; ++r) {
            int row = tile * 16 + r + 8 * hi;
            if (row < TOPN) rank[b * TOPN + row] = (int)(acc[r] + 0.5f);
        }
    }
}

// ---------------------------------------------------------------------------
// Kernel 5: scatter by rank. rank is a permutation of 0..299 so each pos slot
// is written exactly once. Neg ROIs (rows 64..127) are zeros per reference.
// ---------------------------------------------------------------------------
__global__ void select_kernel(const float* __restrict__ nmsb,
                              const int* __restrict__ rank,
                              const int* __restrict__ maxgt,
                              float* __restrict__ roi_out,
                              int* __restrict__ gtidx_out) {
    const int b = blockIdx.x;
    const int t = threadIdx.x;
    if (t < TOPN) {
        int r = rank[b * TOPN + t];
        if (r < NPOS) {
            ((float4*)roi_out)[b * 128 + r] = ((const float4*)nmsb)[b * TOPN + t];
            gtidx_out[b * NPOS + r] = maxgt[b * TOPN + t];
        }
    } else if (t < TOPN + NPOS) {
        int slot = NPOS + (t - TOPN);
        ((float4*)roi_out)[b * 128 + slot] = float4{0.f, 0.f, 0.f, 0.f};
    }
}

// ---------------------------------------------------------------------------
extern "C" void kernel_launch(void* const* d_in, const int* in_sizes, int n_in,
                              void* d_out, int out_size, void* d_ws, size_t ws_size,
                              hipStream_t stream) {
    const float* deltas  = (const float*)d_in[0];  // (32,90000,4)
    const float* labels  = (const float*)d_in[1];  // (32,90000,1)
    const float* anchors = (const float*)d_in[2];  // (32,90000,4)
    const float* gt      = (const float*)d_in[3];  // (32,20,4)

    float* ws     = (float*)d_ws;
    float* dec    = ws;                                   // 11,520,000 f
    float* sc     = ws + 11520000;                        //  2,880,000 f (mutable)
    float* nmsb   = ws + 14400000;                        //     38,400 f
    float* merged = ws + 14438400;                        //      9,600 f
    int*   maxgt  = (int*)(ws + 14448000);                //      9,600 i
    int*   rank   = (int*)(ws + 14457600);                //      9,600 i

    float* roi    = (float*)d_out;                        // (32,128,4) f32
    int*   gtidx  = (int*)((float*)d_out + 32 * 128 * 4); // (32,64) i32

    decode_kernel<<<(BATCH * ANUM + 255) / 256, 256, 0, stream>>>(
        anchors, deltas, labels, dec, sc);
    nms_kernel<<<BATCH, 1024, 0, stream>>>(dec, sc, nmsb);
    iou_gt_kernel<<<BATCH, 320, 0, stream>>>(nmsb, gt, merged, maxgt);
    rank_kernel<<<dim3(19, BATCH), 32, 0, stream>>>(merged, rank);
    select_kernel<<<BATCH, 384, 0, stream>>>(nmsb, rank, maxgt, roi, gtidx);
}
